// M_PosAttention3d_36919538876800
// MI455X (gfx1250) — matrix operand order repaired and checked by reading, not compiled, not run on hardware
//
#include <hip/hip_runtime.h>

// ---------------------------------------------------------------------------
// M_PosAttention3d on MI455X (gfx1250, wave32, WMMA f32_16x16x32_f16)
//
// N=2, IN_CH=64, DK=256, DV=128, NH=8, DKH=32, DVH=16, L=14^3=2744
//
// Roofline: materializing the 16 (2744x2744) attention matrices would cost
// ~1 GB of HBM traffic (>192MB L2) ~ 80us @ 23.3TB/s, vastly more than the
// ~24 GFLOP of compute on WMMA.  So: fused flash attention, implicit-GEMM
// conv via a one-shot im2col, and all operands stored K-contiguous so every
// WMMA operand load is an aligned b128 vector load.
// ---------------------------------------------------------------------------

typedef __attribute__((ext_vector_type(16))) _Float16 v16h;
typedef __attribute__((ext_vector_type(8)))  _Float16 v8h;
typedef __attribute__((ext_vector_type(8)))  float    v8f;

#define WMMA16(A, B, C) __builtin_amdgcn_wmma_f32_16x16x32_f16( \
    false, (A), false, (B), (short)0, (C), false, false)

#define CAT16(LO, HI) __builtin_shufflevector((LO), (HI), \
    0, 1, 2, 3, 4, 5, 6, 7, 8, 9, 10, 11, 12, 13, 14, 15)

#define NBATCH 2
#define INCH   64
#define DKC    256
#define DVC    128
#define NHEAD  8
#define DKH    32
#define DVH    16
#define DSP    14
#define LSP    2744            // 14^3
#define LT     172             // ceil(L/16)
#define KCONV  1728            // 64*27
#define KSTEPS 54              // 1728/32
#define KBLK   86              // ceil(L/32)
#define KPAD   2752            // KBLK*32 (padded key dim)
#define NEG_BIG (-1.0e30f)
// q scale 32^-0.5 with log2(e) folded in so softmax uses raw v_exp_f32 (2^x)
#define QSCALE  (0.17677669529663687f * 1.4426950408889634f)

// ---------------------------------------------------------------------------
// Kernel 1: f32->f16 weight conversion + zero-fill of padded key regions
// ---------------------------------------------------------------------------
__global__ void prep_kernel(const float* __restrict__ qkv_w,
                            const float* __restrict__ out_w,
                            _Float16* __restrict__ wh,
                            _Float16* __restrict__ owh,
                            _Float16* __restrict__ Kt,
                            _Float16* __restrict__ Vt) {
    const int NW = 640 * KCONV;            // 1105920
    const int NO = DVC * DVC;              // 16384
    int i = blockIdx.x * blockDim.x + threadIdx.x;
    if (i < NW) wh[i]  = (_Float16)qkv_w[i];
    if (i < NO) owh[i] = (_Float16)out_w[i];
    // zero-fill Kt pad keys [2744,2752): 16 nh * 8 keys * 32 ch = 4096
    if (i < 4096) {
        int d  = i & 31;
        int ky = LSP + ((i >> 5) & 7);
        int nh = i >> 8;
        Kt[((size_t)nh * KPAD + ky) * DKH + d] = (_Float16)0.0f;
    }
    // zero-fill Vt pad keys: 16 nh * 16 v * 8 keys = 2048
    if (i < 2048) {
        int ky = LSP + (i & 7);
        int v  = (i >> 3) & 15;
        int nh = i >> 7;
        Vt[((size_t)nh * DVH + v) * KPAD + ky] = (_Float16)0.0f;
    }
}

// ---------------------------------------------------------------------------
// Kernel 2: im2col (f32 -> f16), layout P[n][ltile][col(16)][k(1728)]
// so the conv B-operand is K-contiguous per lane.  Pad cols/halo -> 0.
// ---------------------------------------------------------------------------
__global__ void im2col_kernel(const float* __restrict__ x,
                              _Float16* __restrict__ P) {
    const long long NP = (long long)NBATCH * LT * 16 * KCONV;  // 9510912
    long long i = (long long)blockIdx.x * blockDim.x + threadIdx.x;
    if (i >= NP) return;
    int k  = (int)(i % KCONV);
    int t  = (int)(i / KCONV);
    int c  = t % 16;  t /= 16;
    int lt = t % LT;
    int n  = t / LT;

    int l = lt * 16 + c;
    float val = 0.0f;
    if (l < LSP) {
        int z   = l / 196;
        int rem = l % 196;
        int yy  = rem / DSP;
        int xx  = rem % DSP;
        int ic  = k / 27;
        int ko  = k % 27;
        int kz  = ko / 9;
        int ky  = (ko % 9) / 3;
        int kx  = ko % 3;
        int iz = z + kz - 1, iy = yy + ky - 1, ix = xx + kx - 1;
        if (iz >= 0 && iz < DSP && iy >= 0 && iy < DSP && ix >= 0 && ix < DSP)
            val = x[(((n * INCH + ic) * DSP + iz) * DSP + iy) * DSP + ix];
    }
    P[i] = (_Float16)val;
}

// ---------------------------------------------------------------------------
// Kernel 3: qkv conv GEMM.  One wave per 16(oc) x 16(spatial) tile.
// All operand loads are aligned b128.  Epilogue adds bias, scales q
// (QSCALE includes log2e), stores K-contiguous f16 layouts:
//   Qt: [n*8+h][qrow][32]   Kt: [n*8+h][key(KPAD)][32]   Vt: [n*8+h][v][KPAD]
// ---------------------------------------------------------------------------
__global__ void conv_qkv_kernel(const _Float16* __restrict__ P,
                                const _Float16* __restrict__ wh,
                                const float*    __restrict__ qkv_b,
                                _Float16* __restrict__ Qt,
                                _Float16* __restrict__ Kt,
                                _Float16* __restrict__ Vt) {
    const int lane = threadIdx.x;
    const int h    = lane >> 4;
    const int r15  = lane & 15;

    int bid = blockIdx.x;                  // n * 40 * LT
    const int lt = bid % LT;  bid /= LT;
    const int ot = bid % 40;
    const int n  = bid / 40;

    const _Float16* wrow = wh + (size_t)(ot * 16 + r15) * KCONV;
    const _Float16* prow = P  + ((size_t)(n * LT + lt) * 16 + r15) * KCONV;

    v8f acc = {};
    for (int s = 0; s < KSTEPS; ++s) {
        // A: weights, row=oc, elem e -> k = s*32 + (e<8 ? 8h+e : 16+8h+e-8)
        v8h alo = *(const v8h*)(wrow + s * 32 + 8 * h);
        v8h ahi = *(const v8h*)(wrow + s * 32 + 16 + 8 * h);
        // B: patch, col=l, elem e -> k = s*32 + (e<8 ? 8h+e : 16+8h+e-8)
        v8h blo = *(const v8h*)(prow + s * 32 + 8 * h);
        v8h bhi = *(const v8h*)(prow + s * 32 + 16 + 8 * h);
        acc = WMMA16(CAT16(alo, ahi), CAT16(blo, bhi), acc);
    }

    const int lidx = lt * 16 + r15;        // D col (N = r15)
    if (lidx < LSP) {
#pragma unroll
        for (int r = 0; r < 8; ++r) {
            int oc = ot * 16 + r + 8 * h;  // D row
            float v = acc[r] + qkv_b[oc];
            if (oc < DKC) {
                int hd = oc >> 5, d = oc & 31;
                Qt[((size_t)(n * NHEAD + hd) * LSP + lidx) * DKH + d] =
                    (_Float16)(v * QSCALE);
            } else if (oc < 2 * DKC) {
                int c = oc - DKC, hd = c >> 5, d = c & 31;
                Kt[((size_t)(n * NHEAD + hd) * KPAD + lidx) * DKH + d] = (_Float16)v;
            } else {
                int c = oc - 2 * DKC, hd = c >> 4, d = c & 15;
                Vt[((size_t)(n * NHEAD + hd) * DVH + d) * KPAD + lidx] = (_Float16)v;
            }
        }
    }
}

// ---------------------------------------------------------------------------
// Kernel 4: fused flash attention.  One wave per (n, head, 16-row q-tile).
// S^T tiles (A = K-tile, B = Q-tile) so softmax reductions are in-lane plus
// one shfl_xor(16).  85 unmasked key blocks + 1 masked tail.  Logits are in
// log2 domain (QSCALE folded), so exp = raw v_exp_f32.
// ---------------------------------------------------------------------------
template <bool MASK>
__device__ __forceinline__ void attn_block(int kbase,
                                           const _Float16* __restrict__ Kb,
                                           const _Float16* __restrict__ Vb,
                                           const v16h& bq, v8f& acc,
                                           float& m_i, float& l_i,
                                           int h, int r15) {
    // A operands: two K tiles (rows = 16 keys, K = 32 chans), all b128 loads
    const _Float16* k0p = Kb + (size_t)(kbase + r15) * DKH;
    const _Float16* k1p = Kb + (size_t)(kbase + 16 + r15) * DKH;
    v16h ak0 = CAT16(*(const v8h*)(k0p + 8 * h), *(const v8h*)(k0p + 16 + 8 * h));
    v16h ak1 = CAT16(*(const v8h*)(k1p + 8 * h), *(const v8h*)(k1p + 16 + 8 * h));
    // V A-operand: row = v (r15), K = key-local
    const _Float16* vp = Vb + (size_t)r15 * KPAD + kbase;
    v16h av = CAT16(*(const v8h*)(vp + 16 * h), *(const v8h*)(vp + 16 * h + 8));

    v8f z = {};
    v8f s0 = WMMA16(ak0, bq, z);           // S^T keys [kbase, kbase+16)
    v8f s1 = WMMA16(ak1, bq, z);           // S^T keys [kbase+16, kbase+32)

    // online softmax: lane owns 16 of 32 keys for its qrow (D: M = r + 8h)
    float sv[16];
    float tmax = NEG_BIG;
#pragma unroll
    for (int r = 0; r < 8; ++r) {
        float a0 = s0[r], a1 = s1[r];
        if (MASK) {
            if (kbase + r + 8 * h >= LSP)      a0 = NEG_BIG;
            if (kbase + 16 + r + 8 * h >= LSP) a1 = NEG_BIG;
        }
        sv[r] = a0; sv[r + 8] = a1;
        tmax = fmaxf(tmax, fmaxf(a0, a1));
    }
    tmax = fmaxf(tmax, __shfl_xor(tmax, 16, 32));
    float m_new = fmaxf(m_i, tmax);
    float alpha = __builtin_amdgcn_exp2f(m_i - m_new);

    float rsum = 0.0f;
    _Float16 ph[16];
#pragma unroll
    for (int i = 0; i < 16; ++i) {
        float p = __builtin_amdgcn_exp2f(sv[i] - m_new);
        rsum += p;
        ph[i] = (_Float16)p;
    }
    rsum += __shfl_xor(rsum, 16, 32);
    l_i = l_i * alpha + rsum;
    m_i = m_new;
    acc = acc * alpha;

    // P^T (D layout) -> B operand via 4 packed dword exchanges with lane^16.
    // h0 lane holds keys {r},{16+r}; h1 lane holds keys {8+r},{24+r}.
    v8h own_lo = { ph[0], ph[1], ph[2],  ph[3],  ph[4],  ph[5],  ph[6],  ph[7]  };
    v8h own_hi = { ph[8], ph[9], ph[10], ph[11], ph[12], ph[13], ph[14], ph[15] };
    union PU { v8h v; int i[4]; };
    PU snd, rcv;
    snd.v = h ? own_lo : own_hi;
#pragma unroll
    for (int t = 0; t < 4; ++t) rcv.i[t] = __shfl_xor(snd.i[t], 16, 32);
    v8h lo8 = h ? rcv.v  : own_lo;         // B elems 0..7:  keys 16h+0..7
    v8h hi8 = h ? own_hi : rcv.v;          // B elems 8..15: keys 16h+8..15
    acc = WMMA16(av, CAT16(lo8, hi8), acc);  // O^T += V * P^T
}

__global__ void attn_kernel(const _Float16* __restrict__ Qt,
                            const _Float16* __restrict__ Kt,
                            const _Float16* __restrict__ Vt,
                            _Float16* __restrict__ Ot) {
    const int lane = threadIdx.x;
    const int h    = lane >> 4;
    const int r15  = lane & 15;

    int bid = blockIdx.x;                  // n * 8 * LT
    const int qt   = bid % LT;  bid /= LT;
    const int head = bid % NHEAD;
    const int n    = bid / NHEAD;

    const int  qrow = qt * 16 + r15;
    const bool qval = qrow < LSP;
    const int  qr   = qval ? qrow : (LSP - 1);

    const _Float16* Kb = Kt + (size_t)(n * NHEAD + head) * KPAD * DKH;
    const _Float16* Vb = Vt + (size_t)(n * NHEAD + head) * DVH * KPAD;

    // B operand: Q tile (held all kernel). col=qrow, elem e -> chan (e<8 ? 8h+e : 16+8h+e-8)
    const _Float16* qb = Qt + ((size_t)(n * NHEAD + head) * LSP + qr) * DKH;
    v16h bq = CAT16(*(const v8h*)(qb + 8 * h), *(const v8h*)(qb + 16 + 8 * h));

    v8f  acc = {};                         // O^T accumulator: M=v, N=qrow
    float m_i = NEG_BIG, l_i = 0.0f;

    for (int j = 0; j < KBLK - 1; ++j) {
        // prefetch next key/value block (global_prefetch_b8)
        __builtin_prefetch(Kb + (size_t)(j * 32 + 32 + r15) * DKH, 0, 3);
        __builtin_prefetch(Vb + (size_t)r15 * KPAD + j * 32 + 32, 0, 3);
        attn_block<false>(j * 32, Kb, Vb, bq, acc, m_i, l_i, h, r15);
    }
    attn_block<true>((KBLK - 1) * 32, Kb, Vb, bq, acc, m_i, l_i, h, r15);

    if (qval) {
        const float rinv = 1.0f / l_i;
        v8h oh;
#pragma unroll
        for (int r = 0; r < 8; ++r) oh[r] = (_Float16)(acc[r] * rinv);
        // D rows v = r + 8h are contiguous channels -> single b128 store
        *(v8h*)(Ot + ((size_t)n * LSP + qrow) * DVC + head * DVH + 8 * h) = oh;
    }
}

// ---------------------------------------------------------------------------
// Kernel 5: 1x1x1 output projection (WMMA GEMM + bias).  out f32 [n][128][L].
// Ot is [n][l][128] so B-operand is K-contiguous.
// ---------------------------------------------------------------------------
__global__ void outproj_kernel(const _Float16* __restrict__ owh,
                               const _Float16* __restrict__ Ot,
                               const float*    __restrict__ out_b,
                               float* __restrict__ out) {
    const int lane = threadIdx.x;
    const int h    = lane >> 4;
    const int r15  = lane & 15;

    int bid = blockIdx.x;                  // n * 8 * LT
    const int lt = bid % LT;  bid /= LT;
    const int ot = bid % 8;
    const int n  = bid / 8;

    const int lidx = lt * 16 + r15;
    const int li   = (lidx < LSP) ? lidx : (LSP - 1);

    const _Float16* arow = owh + (size_t)(ot * 16 + r15) * DVC;
    const _Float16* brow = Ot + ((size_t)n * LSP + li) * DVC;

    v8f acc = {};
#pragma unroll
    for (int s = 0; s < 4; ++s) {          // K = 128 channels, 4 WMMA steps
        v8h alo = *(const v8h*)(arow + s * 32 + 8 * h);
        v8h ahi = *(const v8h*)(arow + s * 32 + 16 + 8 * h);
        v8h blo = *(const v8h*)(brow + s * 32 + 8 * h);
        v8h bhi = *(const v8h*)(brow + s * 32 + 16 + 8 * h);
        acc = WMMA16(CAT16(alo, ahi), CAT16(blo, bhi), acc);
    }

    if (lidx < LSP) {
#pragma unroll
        for (int r = 0; r < 8; ++r) {
            int oc = ot * 16 + r + 8 * h;
            out[((size_t)n * DVC + oc) * LSP + lidx] = acc[r] + out_b[oc];
        }
    }
}

// ---------------------------------------------------------------------------
// Host-side launcher
// ---------------------------------------------------------------------------
extern "C" void kernel_launch(void* const* d_in, const int* in_sizes, int n_in,
                              void* d_out, int out_size, void* d_ws, size_t ws_size,
                              hipStream_t stream) {
    const float* x      = (const float*)d_in[0];
    const float* qkv_w  = (const float*)d_in[1];
    const float* qkv_b  = (const float*)d_in[2];
    const float* out_w  = (const float*)d_in[3];
    const float* out_b  = (const float*)d_in[4];
    float*       out    = (float*)d_out;

    // Workspace carve-up (256B aligned)
    char* ws = (char*)d_ws;
    size_t off = 0;
    auto carve = [&](size_t bytes) {
        char* p = ws + off;
        off += (bytes + 255) & ~(size_t)255;
        return p;
    };
    _Float16* wh  = (_Float16*)carve((size_t)640 * KCONV * 2);                 //  2.21 MB
    _Float16* owh = (_Float16*)carve((size_t)DVC * DVC * 2);                   //  32 KB
    _Float16* Pim = (_Float16*)carve((size_t)NBATCH * LT * 16 * KCONV * 2);    // 19.02 MB
    _Float16* Qt  = (_Float16*)carve((size_t)NBATCH * NHEAD * LSP * DKH * 2);  //  2.81 MB
    _Float16* Kt  = (_Float16*)carve((size_t)NBATCH * NHEAD * KPAD * DKH * 2); //  2.82 MB
    _Float16* Vt  = (_Float16*)carve((size_t)NBATCH * NHEAD * DVH * KPAD * 2); //  1.41 MB
    _Float16* Ot  = (_Float16*)carve((size_t)NBATCH * LSP * DVC * 2);          //  1.40 MB
    (void)ws_size; (void)in_sizes; (void)n_in; (void)out_size;

    // 1. weight conversion + pad-region zero fill
    {
        int nmax = 640 * KCONV;
        prep_kernel<<<(nmax + 255) / 256, 256, 0, stream>>>(qkv_w, out_w, wh, owh, Kt, Vt);
    }
    // 2. im2col (f32 -> f16, halo zeros baked in)
    {
        long long np = (long long)NBATCH * LT * 16 * KCONV;
        im2col_kernel<<<(int)((np + 255) / 256), 256, 0, stream>>>(x, Pim);
    }
    // 3. qkv conv GEMM: one wave per 16x16 tile
    conv_qkv_kernel<<<NBATCH * 40 * LT, 32, 0, stream>>>(Pim, wh, qkv_b, Qt, Kt, Vt);
    // 4. fused flash attention: one wave per (n, head, q-tile)
    attn_kernel<<<NBATCH * NHEAD * LT, 32, 0, stream>>>(Qt, Kt, Vt, Ot);
    // 5. output projection + bias
    outproj_kernel<<<NBATCH * 8 * LT, 32, 0, stream>>>(owh, Ot, out_b, out);
}
